// voMoE_71605694759038
// MI455X (gfx1250) — matrix operand order repaired and checked by reading, not compiled, not run on hardware
//
#include <hip/hip_runtime.h>
#include <hip/hip_bf16.h>

// ---------------- types ----------------
typedef __attribute__((ext_vector_type(16))) __bf16  v16bf;
typedef __attribute__((ext_vector_type(8)))  float   v8f;
typedef __attribute__((ext_vector_type(4)))  unsigned u32x4;

// ---------------- problem constants ----------------
constexpr int B_ = 4, S_ = 2048, H_ = 1024, E_ = 8;
constexpr int NT = B_ * S_;          // 8192 tokens
constexpr int CAP = NT;              // per-expert list capacity

// workspace layout (bytes)
constexpr size_t OFF_TOK = 1024;
constexpr size_t OFF_WGT = OFF_TOK + (size_t)E_ * CAP * 4;
constexpr size_t OFF_XB  = OFF_WGT + (size_t)E_ * CAP * 4;
constexpr size_t OFF_WB  = OFF_XB  + (size_t)NT * H_ * 2;

static __device__ __forceinline__ unsigned short f2bf(float f) {
  unsigned u = __float_as_uint(f);
  unsigned r = u + 0x7FFFu + ((u >> 16) & 1u);
  return (unsigned short)(r >> 16);
}

// ================= Kernel 0: convert expert_w to bf16, zero counters =================
__global__ __launch_bounds__(256) void moe_convert_w(const float* __restrict__ w,
                                                     unsigned short* __restrict__ wb,
                                                     int* __restrict__ counts) {
  const int n = E_ * H_ * H_;
  int idx = blockIdx.x * blockDim.x + threadIdx.x;
  if (idx < E_) counts[idx] = 0;
  for (int i = idx; i < n; i += gridDim.x * blockDim.x) wb[i] = f2bf(w[i]);
}

// ================= Kernel 1: router (one wave per token) =================
__global__ __launch_bounds__(256) void moe_router(const float* __restrict__ x,
                                                  const float* __restrict__ rw,
                                                  const float* __restrict__ rb,
                                                  const float* __restrict__ eb,
                                                  float* __restrict__ out,
                                                  unsigned short* __restrict__ xb,
                                                  int* __restrict__ counts,
                                                  int* __restrict__ tok_list,
                                                  float* __restrict__ w_list) {
  __shared__ float s_rw[E_ * H_];                       // 32 KB
  const int tid = threadIdx.x;
  for (int i = tid; i < E_ * H_; i += 256) s_rw[i] = rw[i];
  __syncthreads();

  const int wave = tid >> 5;                            // 8 waves / block
  const int lane = tid & 31;
  const int t    = blockIdx.x * 8 + wave;               // 1024 blocks * 8 = 8192 tokens
  const float* xr = x + (size_t)t * H_;

  float acc[E_];
#pragma unroll
  for (int e = 0; e < E_; ++e) acc[e] = 0.f;

#pragma unroll 4
  for (int i = 0; i < H_ / 32; ++i) {
    const int h = i * 32 + lane;
    const float xv = xr[h];
    xb[(size_t)t * H_ + h] = f2bf(xv);                  // fused bf16 conversion of x
#pragma unroll
    for (int e = 0; e < E_; ++e) acc[e] += xv * s_rw[e * H_ + h];
  }
  // wave32 butterfly reduce -> every lane holds the full sums
#pragma unroll
  for (int off = 16; off > 0; off >>= 1)
#pragma unroll
    for (int e = 0; e < E_; ++e) acc[e] += __shfl_xor(acc[e], off, 32);

  float s[E_];
#pragma unroll
  for (int e = 0; e < E_; ++e) s[e] = acc[e] + rb[e];

  // top-2 (ties -> lower index, matching jax.lax.top_k)
  int e0 = 0;
#pragma unroll
  for (int e = 1; e < E_; ++e) if (s[e] > s[e0]) e0 = e;
  int e1 = (e0 == 0) ? 1 : 0;
#pragma unroll
  for (int e = 0; e < E_; ++e) if (e != e0 && s[e] > s[e1]) e1 = e;

  // renormalized top-2 softmax weights: softmax denominators cancel
  const float w0 = 1.f / (1.f + __expf(s[e1] - s[e0]));
  const float w1 = 1.f - w0;

  if (lane == 0) {
    int p0 = atomicAdd(&counts[e0], 1);
    tok_list[e0 * CAP + p0] = t;  w_list[e0 * CAP + p0] = w0;
    int p1 = atomicAdd(&counts[e1], 1);
    tok_list[e1 * CAP + p1] = t;  w_list[e1 * CAP + p1] = w1;
  }

  // initialize output with weighted biases (accumulator base for the GEMM scatter)
  const float* b0 = eb + (size_t)e0 * H_;
  const float* b1 = eb + (size_t)e1 * H_;
  float* orow = out + (size_t)t * H_;
#pragma unroll 4
  for (int i = 0; i < H_ / 32; ++i) {
    const int d = i * 32 + lane;
    orow[d] = w0 * b0[d] + w1 * b1[d];
  }
}

// ================= Kernel 2: grouped GEMM via bf16 WMMA, 2x4 register blocking =================
// grid: (4 col-groups of 256, 256 M-tiles of 32, 8 experts); block: 128 threads = 4 waves.
// Each wave computes a 32x64 C block (2x4 tiles of 16x16) over K=1024:
// per K-step of 32: 2 A + 4 B fragment loads (12 b128) feed 8 WMMAs -> 1.5 loads/WMMA.
__global__ __launch_bounds__(128) void moe_gemm(const unsigned short* __restrict__ xb,
                                                const unsigned short* __restrict__ wb,
                                                const int* __restrict__ counts,
                                                const int* __restrict__ tok_list,
                                                const float* __restrict__ w_list,
                                                float* __restrict__ out) {
  const int e   = blockIdx.z;
  const int cnt = counts[e];
  const int m0  = blockIdx.y * 32;
  if (m0 >= cnt) return;                                // block-uniform -> EXEC stays all-ones

  __shared__ int   s_tok[32];
  __shared__ float s_w[32];
  const int tid = threadIdx.x;
  if (tid < 32) {
    const int idx = m0 + tid;
    if (idx < cnt) { s_tok[tid] = tok_list[e * CAP + idx]; s_w[tid] = w_list[e * CAP + idx]; }
    else           { s_tok[tid] = 0;                       s_w[tid] = 0.f; } // padded rows add 0
  }
  __syncthreads();

  const int wave = tid >> 5;
  const int lane = tid & 31;
  const int hh   = lane >> 4;                           // 0: lanes 0-15, 1: lanes 16-31
  const int ln16 = lane & 15;
  const int colbase = blockIdx.x * 256 + wave * 64;     // this wave's 64 output features

  // A rows: fragment mi covers tokens s_tok[mi*16 .. mi*16+15]
  const unsigned short* xr[2];
  xr[0] = xb + (size_t)s_tok[ln16]      * H_;
  xr[1] = xb + (size_t)s_tok[16 + ln16] * H_;
  // B cols: fragment ni covers output features colbase + ni*16 .. +15 (W is [d][h], k contiguous)
  const unsigned short* wc[4];
#pragma unroll
  for (int ni = 0; ni < 4; ++ni)
    wc[ni] = wb + ((size_t)e << 20) + (size_t)(colbase + ni * 16 + ln16) * H_;

  v8f c[2][4] = {};
#pragma unroll 2
  for (int ks = 0; ks < H_ / 32; ++ks) {
    const int k0 = ks * 32;
    union { u32x4 q[2]; v16bf v; } a[2], b[4];
#pragma unroll
    for (int mi = 0; mi < 2; ++mi) {
      // A 16x32 bf16 fragment (ISA layout): elems0-7 <- K=k0+hh*8.. ; elems8-15 <- K=k0+16+hh*8..
      a[mi].q[0] = *(const u32x4*)(xr[mi] + k0 + hh * 8);
      a[mi].q[1] = *(const u32x4*)(xr[mi] + k0 + 16 + hh * 8);
    }
#pragma unroll
    for (int ni = 0; ni < 4; ++ni) {
      // B 32x16 bf16 fragment: elems0-7 <- K=k0+hh*8.. ; elems8-15 <- K=k0+16+hh*8..
      b[ni].q[0] = *(const u32x4*)(wc[ni] + k0 + hh * 8);
      b[ni].q[1] = *(const u32x4*)(wc[ni] + k0 + 16 + hh * 8);
    }
#pragma unroll
    for (int mi = 0; mi < 2; ++mi)
#pragma unroll
      for (int ni = 0; ni < 4; ++ni)
        c[mi][ni] = __builtin_amdgcn_wmma_f32_16x16x32_bf16(false, a[mi].v, false, b[ni].v,
                                                            (short)0, c[mi][ni], false, false);
  }

  // Epilogue: C tile layout -> VGPR v holds M = v + hh*8, N = ln16.
  // Hoist per-row output pointers + weights so the address chain isn't redone per atomic.
  float*   rowp[2][8];
  float    rwgt[2][8];
#pragma unroll
  for (int mi = 0; mi < 2; ++mi)
#pragma unroll
    for (int v = 0; v < 8; ++v) {
      const int m = mi * 16 + v + hh * 8;
      rowp[mi][v] = out + (size_t)s_tok[m] * H_ + colbase + ln16;
      rwgt[mi][v] = s_w[m];
    }
#pragma unroll
  for (int mi = 0; mi < 2; ++mi)
#pragma unroll
    for (int ni = 0; ni < 4; ++ni)
#pragma unroll
      for (int v = 0; v < 8; ++v)
        atomicAdd(rowp[mi][v] + ni * 16, c[mi][ni][v] * rwgt[mi][v]);
}

// ================= launcher =================
extern "C" void kernel_launch(void* const* d_in, const int* in_sizes, int n_in,
                              void* d_out, int out_size, void* d_ws, size_t ws_size,
                              hipStream_t stream) {
  const float* x   = (const float*)d_in[0];
  const float* rw  = (const float*)d_in[1];
  const float* rb  = (const float*)d_in[2];
  const float* ew  = (const float*)d_in[3];
  const float* eb  = (const float*)d_in[4];
  float* out = (float*)d_out;

  char* ws = (char*)d_ws;
  int*            counts   = (int*)ws;
  int*            tok_list = (int*)(ws + OFF_TOK);
  float*          w_list   = (float*)(ws + OFF_WGT);
  unsigned short* xb       = (unsigned short*)(ws + OFF_XB);
  unsigned short* wb       = (unsigned short*)(ws + OFF_WB);

  // K0: expert_w f32 -> bf16 (stays resident in 192MB L2), zero expert counters
  moe_convert_w<<<2048, 256, 0, stream>>>(ew, wb, counts);
  // K1: router + top-2 + bias init + x f32 -> bf16
  moe_router<<<NT / 8, 256, 0, stream>>>(x, rw, rb, eb, out, xb, counts, tok_list, w_list);
  // K2: grouped GEMM with 2x4 register blocking, weighted scatter-add
  moe_gemm<<<dim3(4, 256, 8), 128, 0, stream>>>(xb, wb, counts, tok_list, w_list, out);
}
